// MultiHeadAttentionDefeasible_17540646437767
// MI455X (gfx1250) — matrix operand order repaired and checked by reading, not compiled, not run on hardware
//
#include <hip/hip_runtime.h>

// Problem constants (match reference)
#define BB 2
#define SS 2048
#define DD 1024
#define HH 16
#define HDm 64
#define MROWS (BB * SS)   // 4096

typedef __bf16 bf16;
typedef __attribute__((ext_vector_type(16))) __bf16 v16bf;
typedef __attribute__((ext_vector_type(8)))  __bf16 v8bf;
typedef __attribute__((ext_vector_type(8)))  float  v8f;
typedef __attribute__((ext_vector_type(4)))  unsigned int u32x4;
typedef __attribute__((ext_vector_type(8)))  unsigned int u32x8;

__device__ __forceinline__ unsigned short f32_to_bf16_bits(float f) {
  unsigned int u = __float_as_uint(f);
  u += 0x7FFFu + ((u >> 16) & 1u);           // round-to-nearest-even
  return (unsigned short)(u >> 16);
}

__global__ __launch_bounds__(256) void cvt_f32_bf16(const float* __restrict__ in,
                                                    unsigned short* __restrict__ out,
                                                    int n) {
  int i = blockIdx.x * blockDim.x + threadIdx.x;
  if (i < n) out[i] = f32_to_bf16_bits(in[i]);
}

// ---- WMMA fragment loaders (CDNA5 16x16x32 bf16 layouts, wave32) ----
// A (16x32, MxK) from row-major src (leading dim ld):
//   lane m = lane&15 ; K-base = (lane>=16 ? 8 : 0)
//   VGPR 0-3 : K = kb..kb+7   (one b128)
//   VGPR 4-7 : K = kb+16..+23 (one b128)
__device__ __forceinline__ v16bf load_frag_a(const bf16* src, int ld, int lane) {
  int m  = lane & 15;
  int kb = (lane >> 4) << 3;
  union { v16bf v; v8bf h[2]; } u;
  u.h[0] = *reinterpret_cast<const v8bf*>(src + (size_t)m * ld + kb);
  u.h[1] = *reinterpret_cast<const v8bf*>(src + (size_t)m * ld + kb + 16);
  return u.v;
}

// B (32x16, KxN) where elem(k,n) = src[n*ld + k]  (i.e. multiply by src^T):
//   lane n = lane&15 ; K-base = (lane>=16 ? 8 : 0); K = kb..kb+7 and kb+16..kb+23
__device__ __forceinline__ v16bf load_frag_bt(const bf16* src, int ld, int lane) {
  int n  = lane & 15;
  int kb = (lane >> 4) << 3;
  union { v16bf v; v8bf h[2]; } u;
  u.h[0] = *reinterpret_cast<const v8bf*>(src + (size_t)n * ld + kb);
  u.h[1] = *reinterpret_cast<const v8bf*>(src + (size_t)n * ld + kb + 16);
  return u.v;
}

__device__ __forceinline__ v8f wmma_bf16(v16bf a, v16bf b, v8f c) {
  return __builtin_amdgcn_wmma_f32_16x16x32_bf16(false, a, false, b, (short)0, c,
                                                 false, false);
}

// ---- GEMM: C(MxN) = X(MxK) @ W(NxK)^T ----
// Block = 8 waves, block tile = 256 (M) x 32 (N). Each wave: 32x32 (2x2 frags).
// The block's 32-row W strip (32 x K bf16 = 64KB) is staged into LDS once via
// the Tensor Data Mover (tensor_load_to_lds, 2D D#), then all waves read
// B-fragments from LDS. A-fragments stream from global (L2-resident).
// mode 0: write bf16, layout (B,H,S,HD)
// mode 1: write bf16, layout (B,H,HD,S)     (transposed heads, for V)
// mode 2: write f32 row-major MxN           (final output)
__global__ __launch_bounds__(256) void gemm_xwt(const bf16* __restrict__ X,
                                                const bf16* __restrict__ W,
                                                void* __restrict__ out,
                                                int M, int N, int K, int mode) {
  __shared__ __align__(16) unsigned short ldsW[32 * DD];   // 32 x K (K<=1024)

  int lane = threadIdx.x & 31;
  int wave = threadIdx.x >> 5;
  int nblks = N >> 5;
  int nblk  = blockIdx.x % nblks;
  int mblk  = blockIdx.x / nblks;
  int n0 = nblk << 5;
  int m0 = (mblk << 8) + (wave << 5);

  // --- TDM: stage W rows [n0, n0+32) x K into LDS (issued by wave 0) ---
  if (wave == 0) {
    unsigned long long gaddr = (unsigned long long)(const void*)(W + (size_t)n0 * K);
    unsigned lds_off = (unsigned)(unsigned long long)(const void*)ldsW;
    u32x4 g0;
    g0[0] = 1u;                                           // count=1, user D#
    g0[1] = lds_off;                                      // lds_addr
    g0[2] = (unsigned)(gaddr & 0xFFFFFFFFu);              // global_addr[31:0]
    g0[3] = (unsigned)((gaddr >> 32) & 0x01FFFFFFu)       // global_addr[56:32]
            | (2u << 30);                                 // type=2 (image)
    u32x8 g1;
    g1[0] = 1u << 16;                                     // data_size=1 -> 2 bytes
    g1[1] = ((unsigned)K & 0xFFFFu) << 16;                // tensor_dim0[15:0]
    g1[2] = ((unsigned)K >> 16) | (32u << 16);            // tdim0[31:16] | tensor_dim1[15:0]=32
    g1[3] = ((unsigned)K & 0xFFFFu) << 16;                // tile_dim0 = K
    g1[4] = 32u;                                          // tile_dim1 = 32
    g1[5] = (unsigned)K;                                  // tensor_dim0_stride[31:0]
    g1[6] = 0u;
    g1[7] = 0u;
    asm volatile("tensor_load_to_lds %0, %1" :: "s"(g0), "s"(g1) : "memory");
    __builtin_amdgcn_s_wait_tensorcnt(0);
  }
  __syncthreads();

  const bf16* Xt0 = X + (size_t)m0 * K;
  const bf16* Xt1 = X + (size_t)(m0 + 16) * K;
  const bf16* Lb  = reinterpret_cast<const bf16*>(ldsW);

  v8f c00 = {}, c01 = {}, c10 = {}, c11 = {};
  for (int k0 = 0; k0 < K; k0 += 32) {
    if (k0 + 64 < K) {
      __builtin_prefetch(Xt0 + k0 + 64, 0, 3);
      __builtin_prefetch(Xt1 + k0 + 64, 0, 3);
    }
    v16bf a0 = load_frag_a(Xt0 + k0, K, lane);
    v16bf a1 = load_frag_a(Xt1 + k0, K, lane);
    v16bf b0 = load_frag_bt(Lb + k0, K, lane);             // LDS rows 0..15
    v16bf b1 = load_frag_bt(Lb + (size_t)16 * K + k0, K, lane); // LDS rows 16..31
    c00 = wmma_bf16(a0, b0, c00);
    c01 = wmma_bf16(a0, b1, c01);
    c10 = wmma_bf16(a1, b0, c10);
    c11 = wmma_bf16(a1, b1, c11);
  }

  int hf = lane >> 4;
  int nl = lane & 15;
#pragma unroll
  for (int i = 0; i < 2; ++i) {
    v8f* crow = (i == 0) ? &c00 : &c10;
#pragma unroll
    for (int j = 0; j < 2; ++j) {
      v8f cc = (j == 0) ? crow[0] : ((i == 0) ? c01 : c11);
      int col = n0 + (j << 4) + nl;
#pragma unroll
      for (int r = 0; r < 8; ++r) {
        int row = m0 + (i << 4) + r + (hf << 3);
        float val = cc[r];
        if (mode == 2) {
          ((float*)out)[(size_t)row * N + col] = val;
        } else {
          int b_ = row / SS, s = row % SS;
          int h  = col >> 6, hd = col & 63;
          unsigned short bits = f32_to_bf16_bits(val);
          size_t addr = (mode == 0)
              ? (((size_t)(b_ * HH + h) * SS + s) * HDm + hd)
              : (((size_t)(b_ * HH + h) * HDm + hd) * SS + s);
          ((unsigned short*)out)[addr] = bits;
        }
      }
    }
  }
}

// ---- def_w[b,h,s] = sum_t sigmoid( (q_def[s] . k_def[t]) / sqrt(HD) ) ----
__global__ __launch_bounds__(256) void defw_kernel(const bf16* __restrict__ qd,
                                                   const bf16* __restrict__ kd,
                                                   float* __restrict__ defw) {
  int lane = threadIdx.x & 31;
  int wave = threadIdx.x >> 5;
  int wid  = blockIdx.x * 8 + wave;
  int bh   = wid / (SS >> 4);
  int s0   = (wid % (SS >> 4)) << 4;
  const bf16* q = qd + ((size_t)bh * SS + s0) * HDm;
  const bf16* k = kd + (size_t)bh * SS * HDm;
  const float inv_sqrt = 0.125f;     // 1/sqrt(64)

  float rs[8];
#pragma unroll
  for (int r = 0; r < 8; ++r) rs[r] = 0.f;

  for (int t0 = 0; t0 < SS; t0 += 16) {
    v8f c = {};
#pragma unroll
    for (int k0 = 0; k0 < HDm; k0 += 32) {
      v16bf a = load_frag_a(q + k0, HDm, lane);
      v16bf b = load_frag_bt(k + (size_t)t0 * HDm + k0, HDm, lane);
      c = wmma_bf16(a, b, c);
    }
#pragma unroll
    for (int r = 0; r < 8; ++r) {
      float x = c[r] * inv_sqrt;
      rs[r] += 1.f / (1.f + __expf(-x));
    }
  }
#pragma unroll
  for (int r = 0; r < 8; ++r)
    for (int off = 1; off < 16; off <<= 1)
      rs[r] += __shfl_xor(rs[r], off, 32);
  if ((lane & 15) == 0) {
    int hf = lane >> 4;
#pragma unroll
    for (int r = 0; r < 8; ++r)
      defw[(size_t)bh * SS + s0 + r + (hf << 3)] = rs[r];
  }
}

// ---- Flash attention with defeasible weights ----
// One wave = 16 query rows. Key tiles of 32. Online softmax in f32.
// q,k: (B,H,S,HD) bf16 ; vt: (B,H,HD,S) bf16 ; attended out: (B,S,D) bf16
__global__ __launch_bounds__(256) void attn_kernel(const bf16* __restrict__ q,
                                                   const bf16* __restrict__ k,
                                                   const bf16* __restrict__ vt,
                                                   const float* __restrict__ defw,
                                                   unsigned short* __restrict__ attended) {
  __shared__ __align__(16) unsigned short ldsP[8][16 * 32];
  int lane = threadIdx.x & 31;
  int wave = threadIdx.x >> 5;
  int wid  = blockIdx.x * 8 + wave;
  int bh   = wid / (SS >> 4);
  int s0   = (wid % (SS >> 4)) << 4;
  int b_   = bh / HH, h = bh % HH;
  const bf16* qt = q  + ((size_t)bh * SS + s0) * HDm;
  const bf16* kb = k  + (size_t)bh * SS * HDm;
  const bf16* vb = vt + (size_t)bh * HDm * SS;
  const float* dw = defw + (size_t)bh * SS;
  const float inv_sqrt = 0.125f;
  int hf = lane >> 4;
  int nl = lane & 15;

  float mrow[8], lrow[8];
  v8f acc[4];
#pragma unroll
  for (int r = 0; r < 8; ++r) { mrow[r] = -1e30f; lrow[r] = 0.f; }
#pragma unroll
  for (int j = 0; j < 4; ++j)
#pragma unroll
    for (int r = 0; r < 8; ++r) acc[j][r] = 0.f;

  unsigned short* myP = ldsP[wave];

  for (int t0 = 0; t0 < SS; t0 += 32) {
    v8f c0 = {}, c1 = {};
#pragma unroll
    for (int k0 = 0; k0 < HDm; k0 += 32) {
      v16bf a  = load_frag_a(qt + k0, HDm, lane);
      v16bf b0 = load_frag_bt(kb + (size_t)t0 * HDm + k0, HDm, lane);
      v16bf b1 = load_frag_bt(kb + (size_t)(t0 + 16) * HDm + k0, HDm, lane);
      c0 = wmma_bf16(a, b0, c0);
      c1 = wmma_bf16(a, b1, c1);
    }
    float w0 = inv_sqrt * dw[t0 + nl];
    float w1 = inv_sqrt * dw[t0 + 16 + nl];
    float mt[8];
#pragma unroll
    for (int r = 0; r < 8; ++r) {
      c0[r] *= w0; c1[r] *= w1;
      mt[r] = fmaxf(c0[r], c1[r]);
    }
#pragma unroll
    for (int r = 0; r < 8; ++r)
      for (int off = 1; off < 16; off <<= 1)
        mt[r] = fmaxf(mt[r], __shfl_xor(mt[r], off, 32));
#pragma unroll
    for (int r = 0; r < 8; ++r) {
      float mn    = fmaxf(mrow[r], mt[r]);
      float alpha = __expf(mrow[r] - mn);
      mrow[r] = mn;
      float p0 = __expf(c0[r] - mn);
      float p1 = __expf(c1[r] - mn);
      float ps = p0 + p1;
      for (int off = 1; off < 16; off <<= 1)
        ps += __shfl_xor(ps, off, 32);
      lrow[r] = lrow[r] * alpha + ps;
#pragma unroll
      for (int j = 0; j < 4; ++j) acc[j][r] *= alpha;
      int row = r + (hf << 3);
      myP[row * 32 + nl]      = f32_to_bf16_bits(p0);
      myP[row * 32 + 16 + nl] = f32_to_bf16_bits(p1);
    }
    // wave-local LDS RAW: P tile must land before A-fragment reload
    asm volatile("s_wait_dscnt 0" ::: "memory");
    v16bf pa = load_frag_a(reinterpret_cast<const bf16*>(myP), 32, lane);
#pragma unroll
    for (int j = 0; j < 4; ++j) {
      v16bf bv = load_frag_bt(vb + (size_t)(j * 16) * SS + t0, SS, lane);
      acc[j] = wmma_bf16(pa, bv, acc[j]);
    }
  }

#pragma unroll
  for (int r = 0; r < 8; ++r) {
    int s = s0 + r + (hf << 3);
    float inv_l = 1.0f / lrow[r];
#pragma unroll
    for (int j = 0; j < 4; ++j) {
      float val = acc[j][r] * inv_l;
      attended[((size_t)b_ * SS + s) * DD + h * HDm + j * 16 + nl] =
          f32_to_bf16_bits(val);
    }
  }
}

extern "C" void kernel_launch(void* const* d_in, const int* in_sizes, int n_in,
                              void* d_out, int out_size, void* d_ws, size_t ws_size,
                              hipStream_t stream) {
  (void)in_sizes; (void)n_in; (void)out_size; (void)ws_size;

  const float* qi  = (const float*)d_in[0];
  const float* ki  = (const float*)d_in[1];
  const float* vi  = (const float*)d_in[2];
  const float* Wq  = (const float*)d_in[3];
  const float* Wqd = (const float*)d_in[4];
  const float* Wk  = (const float*)d_in[5];
  const float* Wkd = (const float*)d_in[6];
  const float* Wv  = (const float*)d_in[7];
  const float* Wo  = (const float*)d_in[8];

  const size_t nX = (size_t)MROWS * DD;   // 4,194,304
  const size_t nW = (size_t)DD * DD;      // 1,048,576

  char* p = (char*)d_ws;
  auto alloc = [&](size_t bytes) -> void* {
    void* r = (void*)p;
    p += (bytes + 255) & ~(size_t)255;
    return r;
  };
  unsigned short* xq  = (unsigned short*)alloc(nX * 2);
  unsigned short* xk  = (unsigned short*)alloc(nX * 2);
  unsigned short* xv  = (unsigned short*)alloc(nX * 2);
  unsigned short* wq  = (unsigned short*)alloc(nW * 2);
  unsigned short* wqd = (unsigned short*)alloc(nW * 2);
  unsigned short* wk  = (unsigned short*)alloc(nW * 2);
  unsigned short* wkd = (unsigned short*)alloc(nW * 2);
  unsigned short* wv  = (unsigned short*)alloc(nW * 2);
  unsigned short* wo  = (unsigned short*)alloc(nW * 2);
  unsigned short* qh  = (unsigned short*)alloc(nX * 2);  // (B,H,S,HD)
  unsigned short* qdh = (unsigned short*)alloc(nX * 2);
  unsigned short* kh  = (unsigned short*)alloc(nX * 2);
  unsigned short* kdh = (unsigned short*)alloc(nX * 2);
  unsigned short* vth = (unsigned short*)alloc(nX * 2);  // (B,H,HD,S)
  unsigned short* att = (unsigned short*)alloc(nX * 2);  // (B,S,D)
  float*          dwf = (float*)alloc((size_t)BB * HH * SS * 4);

  // f32 -> bf16 conversions
  cvt_f32_bf16<<<(int)((nX + 255) / 256), 256, 0, stream>>>(qi, xq, (int)nX);
  cvt_f32_bf16<<<(int)((nX + 255) / 256), 256, 0, stream>>>(ki, xk, (int)nX);
  cvt_f32_bf16<<<(int)((nX + 255) / 256), 256, 0, stream>>>(vi, xv, (int)nX);
  cvt_f32_bf16<<<(int)((nW + 255) / 256), 256, 0, stream>>>(Wq,  wq,  (int)nW);
  cvt_f32_bf16<<<(int)((nW + 255) / 256), 256, 0, stream>>>(Wqd, wqd, (int)nW);
  cvt_f32_bf16<<<(int)((nW + 255) / 256), 256, 0, stream>>>(Wk,  wk,  (int)nW);
  cvt_f32_bf16<<<(int)((nW + 255) / 256), 256, 0, stream>>>(Wkd, wkd, (int)nW);
  cvt_f32_bf16<<<(int)((nW + 255) / 256), 256, 0, stream>>>(Wv,  wv,  (int)nW);
  cvt_f32_bf16<<<(int)((nW + 255) / 256), 256, 0, stream>>>(Wo,  wo,  (int)nW);

  // GEMMs: block tile 256(M) x 32(N) -> (4096/256)*(1024/32) = 512 blocks
  const int gemmBlocks = (MROWS >> 8) * (DD >> 5);
  gemm_xwt<<<gemmBlocks, 256, 0, stream>>>((const bf16*)xq, (const bf16*)wq,  qh,  MROWS, DD, DD, 0);
  gemm_xwt<<<gemmBlocks, 256, 0, stream>>>((const bf16*)xq, (const bf16*)wqd, qdh, MROWS, DD, DD, 0);
  gemm_xwt<<<gemmBlocks, 256, 0, stream>>>((const bf16*)xk, (const bf16*)wk,  kh,  MROWS, DD, DD, 0);
  gemm_xwt<<<gemmBlocks, 256, 0, stream>>>((const bf16*)xk, (const bf16*)wkd, kdh, MROWS, DD, DD, 0);
  gemm_xwt<<<gemmBlocks, 256, 0, stream>>>((const bf16*)xv, (const bf16*)wv,  vth, MROWS, DD, DD, 1);

  // def_w then flash attention: waves = B*H*(S/16) = 4096 -> 512 blocks
  const int attnBlocks = (BB * HH * (SS >> 4)) / 8;
  defw_kernel<<<attnBlocks, 256, 0, stream>>>((const bf16*)qdh, (const bf16*)kdh, dwf);
  attn_kernel<<<attnBlocks, 256, 0, stream>>>((const bf16*)qh, (const bf16*)kh,
                                              (const bf16*)vth, dwf, att);

  // final projection -> f32 output (B,S,D)
  gemm_xwt<<<gemmBlocks, 256, 0, stream>>>((const bf16*)att, (const bf16*)wo,
                                           d_out, MROWS, DD, DD, 2);
}
